// DPR_3753801417615
// MI455X (gfx1250) — compile-verified, matrix operand order checked by reading, not run on hardware
//
#include <hip/hip_runtime.h>
#include <hip/hip_bf16.h>

// ---------------------------------------------------------------------------
// Problem constants
// ---------------------------------------------------------------------------
#define FEAT_DIM 192
#define PTS      2048
#define PROJ     512
#define BATCH    64
#define NSUP     5
#define WAY      6
#define LN_EPS   1e-5f
// 1/sqrt(192)
#define INV_TEMP 0.07216878364870323f

typedef __attribute__((ext_vector_type(16))) __bf16        v16bf;
typedef __attribute__((ext_vector_type(8)))  float         v8f;
typedef __attribute__((ext_vector_type(4)))  unsigned int  v4u;
typedef __attribute__((ext_vector_type(8)))  unsigned int  v8u;

// Workspace layout (bytes)
#define OFF_SUP  ((size_t)0)                       // 6*192*2048 f32  = 9437184 B
#define OFF_QBF  ((size_t)9437184)                 // 64*512*192 bf16 = 12582912 B (fragment-tiled)
#define OFF_KBF  ((size_t)22020096)                // 6*512*192 bf16  = 1179648 B (fragment-tiled)
#define OFF_V    ((size_t)23199744)                // 64*6*192 f32    = 294912 B
#define OFF_PRE  ((size_t)23494656)                // 64*6*192 f32    = 294912 B

__device__ __forceinline__ __bf16 f2bf(float f) { return (__bf16)f; }

// ---------------------------------------------------------------------------
// TDM: 1-D contiguous tile copy, global -> LDS, via tensor descriptor (D#).
// Descriptor bit placement per CDNA5 ISA:
//   g0: [1:0]=count=1, [63:32]=lds_addr, [120:64]=global_addr, [127:126]=type=2
//   g1: [17:16]=data_size(3 -> 8B units), [79:48]=tensor_dim0,
//       [111:80]=tensor_dim1=1, [127:112]=tile_dim0, [143:128]=tile_dim1=1,
//       [207:160]=tensor_dim0_stride
// nbytes must be a multiple of 8 and nbytes/8 < 65536.
// Issued wave-uniform; tracked with TENSORcnt.
// ---------------------------------------------------------------------------
__device__ __forceinline__ void tdm_load_1d(unsigned int lds_off,
                                            const void* gsrc,
                                            unsigned int nbytes)
{
    unsigned long long ga = (unsigned long long)(size_t)gsrc;
    unsigned int units = nbytes >> 3;              // 8-byte units

    v4u g0;
    g0[0] = 1u;                                    // count=1, user descriptor
    g0[1] = lds_off;                               // lds_addr (bytes)
    g0[2] = (unsigned int)(ga & 0xffffffffu);      // global_addr[31:0]
    g0[3] = (unsigned int)((ga >> 32) & 0x01ffffffu) | (2u << 30); // addr hi | type=2

    v8u g1;
    g1[0] = (3u << 16);                            // workgroup_mask=0, data_size=8B
    g1[1] = (units & 0xffffu) << 16;               // tensor_dim0[15:0]  -> bits 63:48
    g1[2] = ((units >> 16) & 0xffffu)              // tensor_dim0[31:16] -> bits 79:64
          | (1u << 16);                            // tensor_dim1 = 1    -> bits 111:80
    g1[3] = (units & 0xffffu) << 16;               // tile_dim0 = units  -> bits 127:112
    g1[4] = 1u;                                    // tile_dim1 = 1, tile_dim2 = 0
    g1[5] = units;                                 // tensor_dim0_stride lo
    g1[6] = 0u;                                    // stride hi / dim1_stride lo
    g1[7] = 0u;

    v4u gz = {0u, 0u, 0u, 0u};

    asm volatile("tensor_load_to_lds %0, %1, %2, %3"
                 :
                 : "s"(g0), "s"(g1), "s"(gz), "s"(gz)
                 : "memory");
}

// ---------------------------------------------------------------------------
// Kernel 0: sup[0] = mean(support), sup[1..5] = support   ([6,192,2048] f32)
// ---------------------------------------------------------------------------
__global__ __launch_bounds__(256) void build_sup_kernel(
    const float* __restrict__ support, float* __restrict__ sup)
{
    const int idx = blockIdx.x * 256 + threadIdx.x;
    const int N = FEAT_DIM * PTS;  // 393216
    if (idx >= N) return;
    float s = 0.f;
#pragma unroll
    for (int w = 0; w < NSUP; ++w) {
        float vv = support[(size_t)w * N + idx];
        s += vv;
        sup[(size_t)(w + 1) * N + idx] = vv;
    }
    sup[idx] = s * (1.0f / NSUP);
}

// ---------------------------------------------------------------------------
// Kernel 0b: v[b,w,j] = prototype[b,w,:] . Wv[j,:] + bv[j]   (384 x 192)
// ---------------------------------------------------------------------------
__global__ __launch_bounds__(192) void compute_v_kernel(
    const float* __restrict__ proto, const float* __restrict__ Wv,
    const float* __restrict__ bv, float* __restrict__ vout)
{
    __shared__ float sp[FEAT_DIM];
    const int bw = blockIdx.x;       // b*6+w
    const int j  = threadIdx.x;
    sp[j] = proto[(size_t)bw * FEAT_DIM + j];
    __syncthreads();
    float a = bv[j];
    const float* wr = Wv + (size_t)j * FEAT_DIM;
    for (int i = 0; i < FEAT_DIM; ++i) a += wr[i] * sp[i];
    vout[(size_t)bw * FEAT_DIM + j] = a;
}

// ---------------------------------------------------------------------------
// Kernel 1: projection GEMM (shared by q and k)
//   logical: O[b, m, d] = alpha * sum_n A[m,n] * Bsrc[b, d, n]
//   A: [512, 2048] f32 (Wq or Wk, L2-resident); Bsrc streamed once (cvt bf16).
// Output is stored in WMMA-fragment-tiled global layouts so the attention
// kernel can stage its tiles as pure contiguous TDM byte copies:
//   mode 0 (q): block(ib = i/16, pt = p/32)[lane][e], 512 bf16 per block,
//               with p = b*8 + m/64, i = (m%64)*192 + d  (the qf "view").
//   mode 1 (k): block(kt = p/32)(nt = j/16)[lane][e],
//               with mi = b*512 + m, p = mi/6, j = (mi%6)*192 + d (kf view).
// Workgroup: 256 thr = 8 waves; tile 128(M) x 192(N); grid = nb*4.
// ---------------------------------------------------------------------------
__global__ __launch_bounds__(256) void proj_gemm_kernel(
    const float* __restrict__ A, const float* __restrict__ Bsrc,
    __bf16* __restrict__ Obf, float alpha, int mode)
{
    // B tile staged in WMMA-fragment order: 12 n-tiles * (32 lanes * 16 el)
    __shared__ __bf16 ldsB[12 * 512];

    const int tid  = threadIdx.x;
    const int lane = tid & 31;
    const int wv   = tid >> 5;
    const int b    = blockIdx.x >> 2;
    const int mt   = blockIdx.x & 3;
    const int m0   = mt * 128 + wv * 16;
    const int mrow = m0 + (lane & 15);
    const int kb   = (lane < 16) ? 0 : 8;   // K-half owned by this lane

    const float* Brow = Bsrc + (size_t)b * FEAT_DIM * PTS;

    v8f acc[12];
    const v8f vzero = {0.f, 0.f, 0.f, 0.f, 0.f, 0.f, 0.f, 0.f};
#pragma unroll
    for (int t = 0; t < 12; ++t) acc[t] = vzero;

    for (int kk = 0; kk < PTS; kk += 32) {
        // prefetch next K chunk of the streamed operand (global_prefetch_b8)
        if (kk + 32 < PTS)
            __builtin_prefetch(Brow + (size_t)(tid & 191) * PTS + kk + 32, 0, 0);

        // cooperative stage of B tile (32 x 192), f32 -> bf16, fragment layout
        for (int e = tid; e < 32 * FEAT_DIM; e += 256) {
            int k = e & 31;           // point index within K tile (contiguous)
            int d = e >> 5;           // feature column
            float vv = Brow[(size_t)d * PTS + kk + k];
            int l2 = (d & 15) + (((k & 15) >= 8) ? 16 : 0);
            int ef = (k & 7) + ((k >= 16) ? 8 : 0);
            ldsB[(d >> 4) * 512 + l2 * 16 + ef] = f2bf(vv);
        }
        __syncthreads();

        // A fragment: 16 f32 per lane, two contiguous runs of 8
        const float* ap = A + (size_t)mrow * PTS + kk + kb;
        float4 a0 = *(const float4*)(ap);
        float4 a1 = *(const float4*)(ap + 4);
        float4 a2 = *(const float4*)(ap + 16);
        float4 a3 = *(const float4*)(ap + 20);
        v16bf af;
        af[0] = f2bf(a0.x);  af[1] = f2bf(a0.y);  af[2] = f2bf(a0.z);  af[3] = f2bf(a0.w);
        af[4] = f2bf(a1.x);  af[5] = f2bf(a1.y);  af[6] = f2bf(a1.z);  af[7] = f2bf(a1.w);
        af[8] = f2bf(a2.x);  af[9] = f2bf(a2.y);  af[10] = f2bf(a2.z); af[11] = f2bf(a2.w);
        af[12] = f2bf(a3.x); af[13] = f2bf(a3.y); af[14] = f2bf(a3.z); af[15] = f2bf(a3.w);

#pragma unroll
        for (int nt = 0; nt < 12; ++nt) {
            v16bf bf = *(const v16bf*)&ldsB[nt * 512 + lane * 16];
            acc[nt] = __builtin_amdgcn_wmma_f32_16x16x32_bf16(
                false, af, false, bf, (short)0, acc[nt], false, false);
        }
        __syncthreads();
    }

    // scatter (scaled) bf16 output into the fragment-tiled layout
#pragma unroll
    for (int nt = 0; nt < 12; ++nt) {
#pragma unroll
        for (int r = 0; r < 8; ++r) {
            int pp = m0 + r + ((lane < 16) ? 0 : 8);   // output row m
            int d  = nt * 16 + (lane & 15);            // output col
            __bf16 val = f2bf(acc[nt][r] * alpha);
            size_t dst;
            if (mode == 0) {
                int p  = b * 8 + (pp >> 6);            // qf row
                int ib = (pp & 63) * 12 + nt;          // i/16
                int k  = p & 31, pt = p >> 5;
                int l2 = (lane & 15) + (((k & 15) >= 8) ? 16 : 0);
                int ef = (k & 7) + ((k >= 16) ? 8 : 0);
                dst = ((size_t)ib * 16 + pt) * 512 + l2 * 16 + ef;
            } else {
                int mi = b * 512 + pp;
                int p  = mi / 6;                       // kf row
                int jw = mi - 6 * p;
                int j  = jw * FEAT_DIM + d;            // kf col
                int kt = p >> 5, k = p & 31;
                int nb = j >> 4;
                int l2 = (j & 15) + (((k & 15) >= 8) ? 16 : 0);
                int ef = (k & 7) + ((k >= 16) ? 8 : 0);
                dst = (size_t)kt * 36864 + (size_t)nb * 512 + l2 * 16 + ef;
            }
            Obf[dst] = val;
        }
    }
}

// ---------------------------------------------------------------------------
// Kernel 2: attn GEMM + blockwise softmax + attn@v, fused, TDM-fed.
//   attn2[i, j] = sum_p qf[p, i] * kf[p, j]   (M=12288, N=1152, K=512)
// Operands arrive pre-permuted in fragment order, so staging is a pure byte
// copy: wave 0 issues tensor_load_to_lds (TDM) for one 16 KB A block (all 16
// K-tiles of this WG) and double-buffered 72 KB B tiles, synchronized with
// s_wait_tensorcnt + workgroup barriers. Inner loop is ds_load_b128 + wmma
// only. attn (56 MB) never touches HBM: softmax over each 192-col block and
// the contraction with v happen in LDS.
// LDS: A 16384 | B0 73728 | B1 73728  = 163840 B (2 WGs / 320KB WGP).
// ---------------------------------------------------------------------------
#define LDS_A_OFF  0u
#define LDS_B0_OFF 16384u
#define LDS_B1_OFF 90112u
#define B_TILE_BYTES 73728u

__global__ __launch_bounds__(192) void attn_softmax_kernel(
    const __bf16* __restrict__ qperm,  // fragment-tiled qf
    const __bf16* __restrict__ kperm,  // fragment-tiled kf
    const float*  __restrict__ vbuf,   // [64, 6, 192] f32
    float* __restrict__ outpre)        // [64, 6, 192] f32
{
    extern __shared__ char dynsmem[];
    const __bf16* ldsA = (const __bf16*)(dynsmem + LDS_A_OFF);
    float*        ldsC = (float*)(dynsmem + LDS_B0_OFF);   // reused after loop

    const int tid  = threadIdx.x;
    const int lane = tid & 31;
    const int wv   = tid >> 5;                  // 0..5 -> column block wv*192
    const int ib   = blockIdx.x;                // i-block: rows i0..i0+15
    const int i0   = ib * 16;

    v8f acc[12];
    const v8f vzero = {0.f, 0.f, 0.f, 0.f, 0.f, 0.f, 0.f, 0.f};
#pragma unroll
    for (int t = 0; t < 12; ++t) acc[t] = vzero;

    // kick off: A block (16 KB, all K-tiles) + first B tile
    if (wv == 0) {
        tdm_load_1d(LDS_A_OFF, qperm + (size_t)ib * 8192, 16384u);
        tdm_load_1d(LDS_B0_OFF, kperm, B_TILE_BYTES);
    }

    for (int t = 0; t < 16; ++t) {
        if (wv == 0) {
            if (t + 1 < 16) {
                // prefetch next B tile into the other buffer, then wait until
                // only that newest TDM op is still outstanding
                tdm_load_1d((t & 1) ? LDS_B0_OFF : LDS_B1_OFF,
                            kperm + (size_t)(t + 1) * 36864, B_TILE_BYTES);
                __builtin_amdgcn_s_wait_tensorcnt(1);
            } else {
                __builtin_amdgcn_s_wait_tensorcnt(0);
            }
        }
        __syncthreads();

        const __bf16* bbase =
            (const __bf16*)(dynsmem + ((t & 1) ? LDS_B1_OFF : LDS_B0_OFF));
        v16bf af = *(const v16bf*)&ldsA[t * 512 + lane * 16];
#pragma unroll
        for (int nt = 0; nt < 12; ++nt) {
            v16bf bf = *(const v16bf*)&bbase[(wv * 12 + nt) * 512 + lane * 16];
            acc[nt] = __builtin_amdgcn_wmma_f32_16x16x32_bf16(
                false, af, false, bf, (short)0, acc[nt], false, false);
        }
        __syncthreads();   // keep consumers behind the next overwrite
    }

    // spill C tile (16 x 1152 f32) to LDS (reusing B0)
#pragma unroll
    for (int nt = 0; nt < 12; ++nt) {
#pragma unroll
        for (int r = 0; r < 8; ++r) {
            int m   = r + ((lane < 16) ? 0 : 8);
            int col = wv * FEAT_DIM + nt * 16 + (lane & 15);
            ldsC[m * 1152 + col] = acc[nt][r];
        }
    }
    __syncthreads();

    // per (row, jw-block) softmax + dot with v; bijection onto (b, w, d1)
    if (tid < 96) {
        int r  = tid / 6;
        int jw = tid - r * 6;
        int i  = i0 + r;
        int bi = i / FEAT_DIM;
        int di = i - bi * FEAT_DIM;
        int m2 = di * 6 + jw;
        int w  = m2 / FEAT_DIM;
        int d1 = m2 - w * FEAT_DIM;

        const float* crow = &ldsC[r * 1152 + jw * FEAT_DIM];
        float mx = -3.402823466e38f;
        for (int t = 0; t < FEAT_DIM; ++t) mx = fmaxf(mx, crow[t]);
        float den = 0.f, av = 0.f;
        const float* vr = vbuf + ((size_t)bi * WAY + w) * FEAT_DIM;
        for (int t = 0; t < FEAT_DIM; ++t) {
            float e = __expf(crow[t] - mx);
            den += e;
            av  += e * vr[t];
        }
        outpre[((size_t)bi * WAY + w) * FEAT_DIM + d1] = av / den;
    }
}

// ---------------------------------------------------------------------------
// Kernel 3: out = LN( Wf @ out_pre + prototype ) * g + b     (384 x 192)
// ---------------------------------------------------------------------------
__global__ __launch_bounds__(192) void fc_ln_kernel(
    const float* __restrict__ pre, const float* __restrict__ Wf,
    const float* __restrict__ proto, const float* __restrict__ g,
    const float* __restrict__ bb, float* __restrict__ out)
{
    __shared__ float spre[FEAT_DIM];
    __shared__ float sx[FEAT_DIM];
    const int bw = blockIdx.x;
    const int o  = threadIdx.x;
    spre[o] = pre[(size_t)bw * FEAT_DIM + o];
    __syncthreads();
    float a = 0.f;
    const float* wr = Wf + (size_t)o * FEAT_DIM;
    for (int i = 0; i < FEAT_DIM; ++i) a += wr[i] * spre[i];
    float x = a + proto[(size_t)bw * FEAT_DIM + o];
    sx[o] = x;
    __syncthreads();
    float mu = 0.f;
    for (int i = 0; i < FEAT_DIM; ++i) mu += sx[i];
    mu *= (1.0f / FEAT_DIM);
    float var = 0.f;
    for (int i = 0; i < FEAT_DIM; ++i) { float d = sx[i] - mu; var += d * d; }
    var *= (1.0f / FEAT_DIM);
    out[(size_t)bw * FEAT_DIM + o] = (x - mu) * rsqrtf(var + LN_EPS) * g[o] + bb[o];
}

// ---------------------------------------------------------------------------
// Host-side launch
// ---------------------------------------------------------------------------
extern "C" void kernel_launch(void* const* d_in, const int* in_sizes, int n_in,
                              void* d_out, int out_size, void* d_ws, size_t ws_size,
                              hipStream_t stream) {
    const float* query   = (const float*)d_in[0];
    const float* support = (const float*)d_in[1];
    const float* proto   = (const float*)d_in[2];
    const float* Wq      = (const float*)d_in[3];
    const float* Wk      = (const float*)d_in[4];
    const float* Wv      = (const float*)d_in[5];
    const float* bv      = (const float*)d_in[6];
    const float* Wf      = (const float*)d_in[7];
    const float* ln_g    = (const float*)d_in[8];
    const float* ln_b    = (const float*)d_in[9];
    float* out = (float*)d_out;

    char* ws = (char*)d_ws;
    float*  sup   = (float*)(ws + OFF_SUP);
    __bf16* qperm = (__bf16*)(ws + OFF_QBF);
    __bf16* kperm = (__bf16*)(ws + OFF_KBF);
    float*  vbuf  = (float*)(ws + OFF_V);
    float*  pre   = (float*)(ws + OFF_PRE);

    // 0) support mean + copy -> sup [6,192,2048]
    build_sup_kernel<<<(FEAT_DIM * PTS + 255) / 256, 256, 0, stream>>>(support, sup);

    // 0b) v = prototype @ Wv.T + bv
    compute_v_kernel<<<BATCH * WAY, FEAT_DIM, 0, stream>>>(proto, Wv, bv, vbuf);

    // 1) q = Wq @ query_b^T (scaled by 1/TEMP), fragment-tiled bf16
    proj_gemm_kernel<<<BATCH * 4, 256, 0, stream>>>(Wq, query, qperm, INV_TEMP, 0);

    // 1b) k = Wk @ sup_w^T, fragment-tiled bf16
    proj_gemm_kernel<<<WAY * 4, 256, 0, stream>>>(Wk, sup, kperm, 1.0f, 1);

    // 2) fused attn GEMM (TDM-fed, double-buffered) + softmax + attn@v
    attn_softmax_kernel<<<12288 / 16, 192, 163840, stream>>>(qperm, kperm, vbuf, pre);

    // 3) FC + residual + LayerNorm
    fc_ln_kernel<<<BATCH * WAY, FEAT_DIM, 0, stream>>>(pre, Wf, proto, ln_g, ln_b, out);
}
